// LSTMCore_20117626815133
// MI455X (gfx1250) — matrix operand order repaired and checked by reading, not compiled, not run on hardware
//
#include <hip/hip_runtime.h>

typedef __attribute__((ext_vector_type(16))) __bf16 v16bf;
typedef __attribute__((ext_vector_type(8)))  __bf16 v8bf;
typedef __attribute__((ext_vector_type(8)))  float  v8f;

#define T_STEPS 512
#define D_IN    128
#define H_DIM   512
#define FOURH   2048
#define KTOT    640      // D_IN + H_DIM (fused K)
#define LDA     648      // padded A-row stride in bf16 (324 dwords % 64 = 4 -> conflict-free)
#define ROWS    16       // batch rows per workgroup
#define NCLS    10
#define HS_LD   (H_DIM + 8)

#define LOG2E   1.44269504088896340736f

// Hardware transcendental fast paths: v_exp_f32 / v_rcp_f32 are TRANS ops that
// co-execute with VALU/WMMA on CDNA5. Saturate correctly for |x| large.
__device__ __forceinline__ float fast_sig(float x) {
    float e = __builtin_amdgcn_exp2f(-LOG2E * x);          // exp(-x)
    return __builtin_amdgcn_rcpf(1.0f + e);
}
__device__ __forceinline__ float fast_tanh(float x) {
    float e = __builtin_amdgcn_exp2f(-2.0f * LOG2E * x);   // exp(-2x)
    return __builtin_fmaf(2.0f, __builtin_amdgcn_rcpf(1.0f + e), -1.0f);
}

// ---------------------------------------------------------------------------
// Prep: pack [W_ih | W_hh] into bf16 row-major [4H][640], bias = b_ih + b_hh
// ---------------------------------------------------------------------------
__global__ void prep_kernel(const float* __restrict__ Wih, const float* __restrict__ Whh,
                            const float* __restrict__ bih, const float* __restrict__ bhh,
                            __bf16* __restrict__ Wc, float* __restrict__ bias)
{
    int idx = blockIdx.x * blockDim.x + threadIdx.x;
    if (idx < FOURH * KTOT) {
        int n = idx / KTOT;
        int k = idx % KTOT;
        float v = (k < D_IN) ? Wih[n * D_IN + k] : Whh[n * H_DIM + (k - D_IN)];
        Wc[idx] = (__bf16)v;
    }
    if (idx < FOURH) bias[idx] = bih[idx] + bhh[idx];
}

// ---------------------------------------------------------------------------
// Persistent LSTM: one workgroup (16 wave32) owns 16 batch rows for all T.
// Wave w owns hidden cols [32w, 32w+32); computes gate tiles i/f/g/o for its
// slice with v_wmma_f32_16x16x32_bf16; c-state & h-sum live in VGPRs in the
// WMMA accumulator layout; h_t is restaged into LDS in A-matrix layout.
// ---------------------------------------------------------------------------
__global__ __launch_bounds__(512) void lstm_kernel(
    const float*  __restrict__ X,
    const __bf16* __restrict__ Wc,
    const float*  __restrict__ bias,
    const float*  __restrict__ Wlin,
    const float*  __restrict__ blin,
    float*        __restrict__ out)
{
    __shared__ __bf16 Abuf[ROWS * LDA];      // [x_t | h_{t-1}] staging, bf16
    __shared__ float  Hs[ROWS * HS_LD];      // h-sum staging for final linear

    const int tid = threadIdx.x;
    const int w   = tid >> 5;                // wave id 0..15
    const int l   = tid & 31;                // lane
    const int lm  = l & 15;                  // lane % 16
    const int hi  = l >> 4;                  // lane half (K-group select)
    const int b0  = blockIdx.x * ROWS;       // first batch row of this WG

    // x-staging assignment: each thread owns one float4 of one row per step
    const int xrow = tid >> 5;
    const int xk4  = (tid & 31) * 4;
    const float* __restrict__ xsrc = X + ((size_t)(b0 + xrow) * T_STEPS) * D_IN + xk4;
    __bf16* __restrict__ xdst = &Abuf[xrow * LDA + xk4];

    // zero A buffer (h region stays zero == h_0)
    for (int i = tid; i < ROWS * LDA; i += 512) Abuf[i] = (__bf16)0.0f;
    __syncthreads();

    // stage x_0
    {
        const float4 x4 = *(const float4*)(xsrc);
        xdst[0] = (__bf16)x4.x; xdst[1] = (__bf16)x4.y;
        xdst[2] = (__bf16)x4.z; xdst[3] = (__bf16)x4.w;
    }

    // per-lane bias preload (bias depends only on N column = lane%16 mapping)
    float bg[4][2];
#pragma unroll
    for (int g = 0; g < 4; ++g)
#pragma unroll
        for (int j = 0; j < 2; ++j)
            bg[g][j] = bias[g * H_DIM + w * 32 + j * 16 + lm];

    v8f c[2], hsum[2];
#pragma unroll
    for (int j = 0; j < 2; ++j)
#pragma unroll
        for (int r = 0; r < 8; ++r) { c[j][r] = 0.0f; hsum[j][r] = 0.0f; }

    const __bf16* arow = &Abuf[lm * LDA];    // A-matrix row M = lane%16

    for (int t = 0; t < T_STEPS; ++t) {
        __syncthreads();                     // [x_t | h_{t-1}] ready in LDS

        // prefetch x_{t+1} into VGPRs now; it flies during the WMMA loop
        float4 xnext;
        if (t + 1 < T_STEPS) xnext = *(const float4*)(xsrc + (size_t)(t + 1) * D_IN);

        v8f acc[4][2];
#pragma unroll
        for (int g = 0; g < 4; ++g)
#pragma unroll
            for (int j = 0; j < 2; ++j)
#pragma unroll
                for (int r = 0; r < 8; ++r) acc[g][j][r] = 0.0f;

#pragma unroll 4
        for (int kk = 0; kk < KTOT; kk += 32) {
            // A fragment (16x32 bf16): lane<16 -> K {0..15}, lane>=16 -> {16..31}
            v8bf alo = *(const v8bf*)(arow + kk + hi * 16);
            v8bf ahi = *(const v8bf*)(arow + kk + hi * 16 + 8);
            v16bf a = __builtin_shufflevector(alo, ahi,
                          0, 1, 2, 3, 4, 5, 6, 7, 8, 9, 10, 11, 12, 13, 14, 15);
#pragma unroll
            for (int g = 0; g < 4; ++g) {
#pragma unroll
                for (int j = 0; j < 2; ++j) {
                    // B fragment (32x16 bf16): lane reads 16 contiguous K at its N column
                    const int col = g * H_DIM + w * 32 + j * 16 + lm;
                    const v16bf bf = *(const v16bf*)(Wc + (size_t)col * KTOT + kk + hi * 16);
                    acc[g][j] = __builtin_amdgcn_wmma_f32_16x16x32_bf16(
                        false, a, false, bf, (short)0, acc[g][j], false, false);
                }
            }
        }
        __syncthreads();                     // all A reads done; safe to restage

        // LSTM cell update, purely elementwise in accumulator layout
        // (bias folded in here instead of accumulator init)
#pragma unroll
        for (int j = 0; j < 2; ++j) {
#pragma unroll
            for (int r = 0; r < 8; ++r) {
                float iv = fast_sig (acc[0][j][r] + bg[0][j]);
                float fv = fast_sig (acc[1][j][r] + bg[1][j]);
                float gv = fast_tanh(acc[2][j][r] + bg[2][j]);
                float ov = fast_sig (acc[3][j][r] + bg[3][j]);
                float cn = fv * c[j][r] + iv * gv;
                c[j][r]  = cn;
                float hv = ov * fast_tanh(cn);
                hsum[j][r] += hv;
                // write h_t back into A staging (row M = r + 8*hi, col 128 + n)
                Abuf[(r + hi * 8) * LDA + D_IN + w * 32 + j * 16 + lm] = (__bf16)hv;
            }
        }

        // stage prefetched x_{t+1}
        if (t + 1 < T_STEPS) {
            xdst[0] = (__bf16)xnext.x; xdst[1] = (__bf16)xnext.y;
            xdst[2] = (__bf16)xnext.z; xdst[3] = (__bf16)xnext.w;
        }
    }

    // dump h-sum (accumulator layout -> row-major LDS)
#pragma unroll
    for (int j = 0; j < 2; ++j)
#pragma unroll
        for (int r = 0; r < 8; ++r)
            Hs[(r + hi * 8) * HS_LD + w * 32 + j * 16 + lm] = hsum[j][r];
    __syncthreads();

    // logits = (h_sum / T) @ W_lin.T + b_lin   (16 rows x 10 classes)
    if (tid < ROWS * NCLS) {
        const int row = tid / NCLS;
        const int cls = tid % NCLS;
        float s = 0.0f;
        for (int k = 0; k < H_DIM; ++k)
            s += Hs[row * HS_LD + k] * Wlin[cls * H_DIM + k];
        out[(b0 + row) * NCLS + cls] = s * (1.0f / (float)T_STEPS) + blin[cls];
    }
}

// ---------------------------------------------------------------------------
extern "C" void kernel_launch(void* const* d_in, const int* in_sizes, int n_in,
                              void* d_out, int out_size, void* d_ws, size_t ws_size,
                              hipStream_t stream)
{
    const float* X    = (const float*)d_in[0];
    const float* Wih  = (const float*)d_in[1];
    const float* Whh  = (const float*)d_in[2];
    const float* bih  = (const float*)d_in[3];
    const float* bhh  = (const float*)d_in[4];
    const float* Wlin = (const float*)d_in[5];
    const float* blin = (const float*)d_in[6];
    float* out = (float*)d_out;

    // workspace: bf16 packed weights [4H][640], then fused bias [4H] fp32
    __bf16* Wc   = (__bf16*)d_ws;
    float*  bias = (float*)((char*)d_ws + (size_t)FOURH * KTOT * sizeof(__bf16));

    const int prep_threads = 256;
    const int prep_blocks  = (FOURH * KTOT + prep_threads - 1) / prep_threads;
    prep_kernel<<<prep_blocks, prep_threads, 0, stream>>>(Wih, Whh, bih, bhh, Wc, bias);

    // 256 batch rows / 16 rows per WG = 16 persistent workgroups of 16 wave32
    lstm_kernel<<<256 / ROWS, 512, 0, stream>>>(X, Wc, bias, Wlin, blin, out);
}
